// ResidualBlock_54047868453484
// MI455X (gfx1250) — matrix operand order repaired and checked by reading, not compiled, not run on hardware
//
#include <hip/hip_runtime.h>
#include <hip/hip_bf16.h>
#include <stdint.h>

// Problem constants (match reference)
#define B_    4
#define D_    1024
#define N_    2048
#define H_    16
#define DH_   64
#define DHID_ 4096
#define SIXD_ (6 * D_)

typedef __attribute__((ext_vector_type(16))) __bf16 v16bf;
typedef __attribute__((ext_vector_type(8)))  float  v8f;

union AFrag { uint32_t u[8]; uint4 q[2]; v16bf v; };

__device__ __forceinline__ uint16_t f2bf(float f) {
  union { float f; uint32_t u; } x; x.f = f;
  uint32_t u = x.u + 0x7FFFu + ((x.u >> 16) & 1u);   // round-to-nearest-even
  return (uint16_t)(u >> 16);
}

// Pack two rows of 8 bf16 (even row a, odd row b) into 8 dwords {a[i], b[i]}
__device__ __forceinline__ void pack8(const uint4 a, const uint4 b, uint32_t* out) {
  const uint32_t* pa = (const uint32_t*)&a;
  const uint32_t* pb = (const uint32_t*)&b;
#pragma unroll
  for (int i = 0; i < 4; ++i) {
    uint32_t lo = pa[i], hi = pb[i];
    out[2 * i]     = (lo & 0xFFFFu) | (hi << 16);
    out[2 * i + 1] = (lo >> 16) | (hi & 0xFFFF0000u);
  }
}

// ---------------------------------------------------------------------------
// f32 -> bf16 conversion (weights)
// ---------------------------------------------------------------------------
__global__ __launch_bounds__(256) void cvt_bf16(const float* __restrict__ in,
                                                uint16_t* __restrict__ out, int n) {
  int i = blockIdx.x * 256 + threadIdx.x;
  if (i < n) out[i] = f2bf(in[i]);
}

// ---------------------------------------------------------------------------
// ada = silu(c) @ w_ada^T + b_ada      (B, 6D)
// ---------------------------------------------------------------------------
__global__ __launch_bounds__(256) void ada_kernel(const float* __restrict__ c,
                                                  const float* __restrict__ w,
                                                  const float* __restrict__ bias,
                                                  float* __restrict__ ada) {
  __shared__ float sc[D_];
  const int b = blockIdx.y;
  for (int i = threadIdx.x; i < D_; i += 256) {
    float v = c[b * D_ + i];
    sc[i] = v / (1.0f + __expf(-v));
  }
  __syncthreads();
  const int o = blockIdx.x * 256 + threadIdx.x;
  const float* wr = w + (size_t)o * D_;
  float acc = 0.0f;
  for (int i = 0; i < D_; ++i) acc += wr[i] * sc[i];
  ada[b * SIXD_ + o] = acc + bias[o];
}

// ---------------------------------------------------------------------------
// channel-LN over D + (1+sc)*x_hat + sh  ->  bf16  (B, D, N)
// ---------------------------------------------------------------------------
__global__ __launch_bounds__(256) void ln_mod(const float* __restrict__ X,
                                              const float* __restrict__ ada,
                                              int scOff, int shOff,
                                              uint16_t* __restrict__ Hb) {
  __shared__ float red[2][8][32];
  const int l = threadIdx.x & 31;
  const int chunk = threadIdx.x >> 5;
  const int n = blockIdx.x * 32 + l;
  const int b = blockIdx.y;
  const float* xb = X + (size_t)b * D_ * N_;
  float s = 0.0f, ss = 0.0f;
  for (int d = chunk * 128; d < chunk * 128 + 128; ++d) {
    float v = xb[(size_t)d * N_ + n];
    s += v; ss += v * v;
  }
  red[0][chunk][l] = s; red[1][chunk][l] = ss;
  __syncthreads();
  s = 0.0f; ss = 0.0f;
  for (int k = 0; k < 8; ++k) { s += red[0][k][l]; ss += red[1][k][l]; }
  const float mu  = s * (1.0f / D_);
  const float var = ss * (1.0f / D_) - mu * mu;
  const float rs  = rsqrtf(var + 1e-5f);
  const float* adab = ada + b * SIXD_;
  for (int d = chunk * 128; d < chunk * 128 + 128; ++d) {
    float sc = 1.0f + adab[scOff + d];
    float sh = adab[shOff + d];
    float v = (xb[(size_t)d * N_ + n] - mu) * rs * sc + sh;
    Hb[((size_t)b * D_ + d) * N_ + n] = f2bf(v);
  }
}

// ---------------------------------------------------------------------------
// WMMA GEMM: out[b,o,n] = sum_c W[o,c]*X[b,c,n] + bias[o]
//   EPI 0: bf16 store    EPI 1: exact GELU -> bf16    EPI 2: res + gate*v (f32)
// Block tile 256x128, K-step 32, 8 waves (4m x 2n), wave = 64x64 (16 WMMA/step).
// Double-buffered LDS: next tile's global loads issue before compute phase.
// ---------------------------------------------------------------------------
template <int EPI>
__global__ __launch_bounds__(256) void gemm_wmma(
    const uint16_t* __restrict__ Wb, const uint16_t* __restrict__ Xb,
    const float* __restrict__ bias, uint16_t* __restrict__ outB,
    float* __restrict__ outF, const float* __restrict__ res,
    const float* __restrict__ ada, int gateOff, int Mdim, int Kdim) {
  __shared__ uint16_t As[2][256][40];   // (m, k) pitch 80B
  __shared__ uint32_t Bp[2][128][20];   // (col, kpair) packed dwords, pitch 80B

  const int tid  = threadIdx.x;
  const int lane = tid & 31;
  const int wave = tid >> 5;
  const int wm   = wave >> 1;        // 0..3 -> 64 rows each
  const int wn   = wave & 1;         // 0..1 -> 64 cols each
  const int half = lane >> 4;
  const int l16  = lane & 15;
  const int bm = blockIdx.y * 256;
  const int bn = blockIdx.x * 128;
  const int bz = blockIdx.z;
  const uint16_t* Xbase = Xb + (size_t)bz * Kdim * N_;

  v8f acc[4][4];
  for (int ms = 0; ms < 4; ++ms)
    for (int ns = 0; ns < 4; ++ns)
      for (int j = 0; j < 8; ++j) acc[ms][ns][j] = 0.0f;

  const int arow = tid;                              // A: one full 32-elem row
  const int bkp  = tid >> 4, bnc = (tid & 15) * 8;   // B: one k-pair, 8 columns

  const uint16_t* agp = Wb + (size_t)(bm + arow) * Kdim;
  const uint16_t* bg0 = Xbase + (size_t)(2 * bkp) * N_ + bn + bnc;

  uint4 sa[4], se, so;
  auto loadStage = [&](int k0) {
#pragma unroll
    for (int q = 0; q < 4; ++q) sa[q] = *(const uint4*)(agp + k0 + q * 8);
    se = *(const uint4*)(bg0 + (size_t)k0 * N_);
    so = *(const uint4*)(bg0 + (size_t)k0 * N_ + N_);
  };
  auto storeStage = [&](int buf) {
#pragma unroll
    for (int q = 0; q < 4; ++q) *(uint4*)&As[buf][arow][q * 8] = sa[q];
    uint32_t pk[8];
    pack8(se, so, pk);
#pragma unroll
    for (int i = 0; i < 8; ++i) Bp[buf][bnc + i][bkp] = pk[i];
  };

  loadStage(0);
  storeStage(0);
  __syncthreads();

  int cur = 0;
  for (int k0 = 32; k0 <= Kdim; k0 += 32) {
    const bool more = k0 < Kdim;
    if (more) loadStage(k0);   // global loads overlap with compute below

    AFrag bf[4];
#pragma unroll
    for (int ns = 0; ns < 4; ++ns) {
      const uint32_t* bp = &Bp[cur][wn * 64 + ns * 16 + l16][half * 8];
      bf[ns].q[0] = *(const uint4*)bp;        // kpairs half*8 .. +3
      bf[ns].q[1] = *(const uint4*)(bp + 4);  // kpairs half*8+4 .. +7
    }
#pragma unroll
    for (int ms = 0; ms < 4; ++ms) {
      AFrag a;
      const uint16_t* ap = &As[cur][wm * 64 + ms * 16 + l16][half * 16];
      a.q[0] = *(const uint4*)ap;             // K = 16*half .. +7
      a.q[1] = *(const uint4*)(ap + 8);       // K = 16*half+8 .. +15
#pragma unroll
      for (int ns = 0; ns < 4; ++ns)
        acc[ms][ns] = __builtin_amdgcn_wmma_f32_16x16x32_bf16(
            false, a.v, false, bf[ns].v, (short)0, acc[ms][ns], false, false);
    }

    if (more) storeStage(cur ^ 1);
    __syncthreads();
    cur ^= 1;
  }

  // Epilogue
#pragma unroll
  for (int ms = 0; ms < 4; ++ms)
#pragma unroll
    for (int ns = 0; ns < 4; ++ns)
#pragma unroll
      for (int j = 0; j < 8; ++j) {
        const int o = bm + wm * 64 + ms * 16 + half * 8 + j;
        const int n = bn + wn * 64 + ns * 16 + l16;
        float v = acc[ms][ns][j] + bias[o];
        const size_t idx = ((size_t)bz * Mdim + o) * N_ + n;
        if (EPI == 0) {
          outB[idx] = f2bf(v);
        } else if (EPI == 1) {
          float g = 0.5f * v * (1.0f + erff(v * 0.70710678118654752f));
          outB[idx] = f2bf(g);
        } else {
          float gate = ada[bz * SIXD_ + gateOff + o];
          outF[idx] = res[idx] + gate * v;
        }
      }
}

// ---------------------------------------------------------------------------
// Flash attention: per (b, h, 128-query tile). 8 waves, 16 queries each.
// Double-buffered K/V tiles. Softmax denominator accumulated via an extra
// WMMA against an all-ones B tile (no cross-lane sum reduction needed).
// ---------------------------------------------------------------------------
__global__ __launch_bounds__(256) void attn_kernel(
    const uint16_t* __restrict__ Qb, const uint16_t* __restrict__ KVb,
    uint16_t* __restrict__ attB) {
  __shared__ uint32_t Kp[2][32][36];     // (m, dpair) packed dwords, pitch 144B
  __shared__ uint16_t Vs[2][64][40];     // (d, m)
  __shared__ uint16_t Ps[8][16][40];     // per-wave P patch (n, m)

  const int tid  = threadIdx.x;
  const int lane = tid & 31;
  const int wave = tid >> 5;
  const int half = lane >> 4;
  const int l16  = lane & 15;
  const int hh = blockIdx.y;
  const int b  = blockIdx.z;
  const int n0 = blockIdx.x * 128 + wave * 16;
  const size_t qch = (size_t)b * D_ + hh * DH_;
  const size_t kch = (size_t)b * (2 * D_) + hh * DH_;
  const size_t vch = kch + D_;

  // Resident Q fragments: A-layout, rows = 16 queries, K = d (two 32-chunks)
  AFrag qa[2];
#pragma unroll
  for (int f = 0; f < 2; ++f) {
    const int row = n0 + l16;
    const int kb = f * 32 + half * 16;
#pragma unroll
    for (int j = 0; j < 4; ++j) {
      int d0 = kb + 2 * j;
      qa[f].u[j] = (uint32_t)Qb[(qch + d0) * N_ + row] |
                   ((uint32_t)Qb[(qch + d0 + 1) * N_ + row] << 16);
      int d1 = kb + 8 + 2 * j;
      qa[f].u[4 + j] = (uint32_t)Qb[(qch + d1) * N_ + row] |
                       ((uint32_t)Qb[(qch + d1 + 1) * N_ + row] << 16);
    }
  }
  AFrag onesB;   // all-ones 32x16 B tile (bf16 1.0 pairs)
#pragma unroll
  for (int j = 0; j < 8; ++j) onesB.u[j] = 0x3F803F80u;

  v8f o[5];      // o[0..3] = output d-tiles, o[4] = softmax denominator
  for (int t = 0; t < 5; ++t)
    for (int j = 0; j < 8; ++j) o[t][j] = 0.0f;
  float mrun[8];
  for (int j = 0; j < 8; ++j) mrun[j] = -1e30f;

  const int vrow = tid >> 2, vq = (tid & 3) * 8;   // V copy: all 256 threads
  const int kdp = tid >> 2, kmc = (tid & 3) * 8;   // K pack: threads 0..127

  uint4 sv, ke, ko;
  auto loadTile = [&](int m0) {
    sv = *(const uint4*)(KVb + (vch + vrow) * N_ + m0 + vq);
    if (tid < 128) {
      ke = *(const uint4*)(KVb + (kch + 2 * kdp) * N_ + m0 + kmc);
      ko = *(const uint4*)(KVb + (kch + 2 * kdp + 1) * N_ + m0 + kmc);
    }
  };
  auto storeTile = [&](int buf) {
    *(uint4*)&Vs[buf][vrow][vq] = sv;
    if (tid < 128) {
      uint32_t pk[8];
      pack8(ke, ko, pk);
#pragma unroll
      for (int i = 0; i < 8; ++i) Kp[buf][kmc + i][kdp] = pk[i];
    }
  };

  loadTile(0);
  storeTile(0);
  __syncthreads();

  int cur = 0;
  for (int m0 = 32; m0 <= N_; m0 += 32) {
    const bool more = m0 < N_;
    if (more) loadTile(m0);   // overlap next tile's HBM latency with compute

    // S tile 16x32 = Q^T K : hoist all 4 K fragments, then 4 WMMAs
    AFrag bk[4];
#pragma unroll
    for (int f = 0; f < 2; ++f) {
      const int dp = f * 16 + half * 8;
      const uint32_t* kl = &Kp[cur][l16][dp];
      bk[2 * f].q[0] = *(const uint4*)kl;
      bk[2 * f].q[1] = *(const uint4*)(kl + 4);
      const uint32_t* kh = &Kp[cur][16 + l16][dp];
      bk[2 * f + 1].q[0] = *(const uint4*)kh;
      bk[2 * f + 1].q[1] = *(const uint4*)(kh + 4);
    }
    v8f sLo, sHi;
    for (int j = 0; j < 8; ++j) { sLo[j] = 0.0f; sHi[j] = 0.0f; }
    sLo = __builtin_amdgcn_wmma_f32_16x16x32_bf16(false, qa[0].v, false, bk[0].v,
                                                  (short)0, sLo, false, false);
    sHi = __builtin_amdgcn_wmma_f32_16x16x32_bf16(false, qa[0].v, false, bk[1].v,
                                                  (short)0, sHi, false, false);
    sLo = __builtin_amdgcn_wmma_f32_16x16x32_bf16(false, qa[1].v, false, bk[2].v,
                                                  (short)0, sLo, false, false);
    sHi = __builtin_amdgcn_wmma_f32_16x16x32_bf16(false, qa[1].v, false, bk[3].v,
                                                  (short)0, sHi, false, false);
#pragma unroll
    for (int j = 0; j < 8; ++j) { sLo[j] *= 0.125f; sHi[j] *= 0.125f; }

    // Row max over 32 keys (butterfly over the 16-lane group holding the row)
    float rmax[8];
#pragma unroll
    for (int j = 0; j < 8; ++j) rmax[j] = fmaxf(sLo[j], sHi[j]);
    for (int off = 1; off < 16; off <<= 1)
#pragma unroll
      for (int j = 0; j < 8; ++j) rmax[j] = fmaxf(rmax[j], __shfl_xor(rmax[j], off, 32));

    float alpha[8], pLo[8], pHi[8];
#pragma unroll
    for (int j = 0; j < 8; ++j) {
      float mn = fmaxf(mrun[j], rmax[j]);
      alpha[j] = __expf(mrun[j] - mn);
      mrun[j] = mn;
      pLo[j] = __expf(sLo[j] - mn);
      pHi[j] = __expf(sHi[j] - mn);
    }
#pragma unroll
    for (int t = 0; t < 5; ++t)
#pragma unroll
      for (int j = 0; j < 8; ++j) o[t][j] *= alpha[j];

    // P (C-layout) -> A-fragment via per-wave LDS patch (same-wave RAW)
#pragma unroll
    for (int j = 0; j < 8; ++j) {
      Ps[wave][j + half * 8][l16]      = f2bf(pLo[j]);
      Ps[wave][j + half * 8][16 + l16] = f2bf(pHi[j]);
    }
    // V fragments (issue early so DS latency overlaps the P readback)
    AFrag bv[4];
#pragma unroll
    for (int t = 0; t < 4; ++t) {
      const uint16_t* vp = &Vs[cur][t * 16 + l16][half * 16];
      bv[t].q[0] = *(const uint4*)vp;
      bv[t].q[1] = *(const uint4*)(vp + 8);
    }
    AFrag pf;
    {
      const uint16_t* pp = &Ps[wave][l16][half * 16];
      pf.q[0] = *(const uint4*)pp;
      pf.q[1] = *(const uint4*)(pp + 8);
    }
    // O += P * V^T ; denominator += P * ones
#pragma unroll
    for (int t = 0; t < 4; ++t)
      o[t] = __builtin_amdgcn_wmma_f32_16x16x32_bf16(false, pf.v, false, bv[t].v,
                                                     (short)0, o[t], false, false);
    o[4] = __builtin_amdgcn_wmma_f32_16x16x32_bf16(false, pf.v, false, onesB.v,
                                                   (short)0, o[4], false, false);

    if (more) storeTile(cur ^ 1);
    __syncthreads();
    cur ^= 1;
  }

  // Normalize and store att (B, D, N) bf16; o[4] holds row denominators
  float inv[8];
#pragma unroll
  for (int j = 0; j < 8; ++j) inv[j] = 1.0f / o[4][j];
#pragma unroll
  for (int t = 0; t < 4; ++t)
#pragma unroll
    for (int j = 0; j < 8; ++j) {
      const int d = hh * DH_ + t * 16 + l16;
      const int n = n0 + j + half * 8;
      attB[((size_t)b * D_ + d) * N_ + n] = f2bf(o[t][j] * inv[j]);
    }
}

// ---------------------------------------------------------------------------
// Host-side orchestration
// ---------------------------------------------------------------------------
extern "C" void kernel_launch(void* const* d_in, const int* in_sizes, int n_in,
                              void* d_out, int out_size, void* d_ws, size_t ws_size,
                              hipStream_t stream) {
  const float* x     = (const float*)d_in[0];
  const float* c     = (const float*)d_in[1];
  const float* w_q   = (const float*)d_in[2];
  const float* b_q   = (const float*)d_in[3];
  const float* w_kv  = (const float*)d_in[4];
  const float* b_kv  = (const float*)d_in[5];
  const float* w_o   = (const float*)d_in[6];
  const float* b_o   = (const float*)d_in[7];
  const float* w_ada = (const float*)d_in[8];
  const float* b_ada = (const float*)d_in[9];
  const float* w_m1  = (const float*)d_in[10];
  const float* b_m1  = (const float*)d_in[11];
  const float* w_m2  = (const float*)d_in[12];
  const float* b_m2  = (const float*)d_in[13];

  char* ws = (char*)d_ws;
  size_t off = 0;
  auto take = [&](size_t bytes) -> char* {
    char* p = ws + off;
    off += (bytes + 255) & ~(size_t)255;
    return p;
  };
  uint16_t* wqB  = (uint16_t*)take((size_t)D_ * D_ * 2);
  uint16_t* wkvB = (uint16_t*)take((size_t)2 * D_ * D_ * 2);
  uint16_t* woB  = (uint16_t*)take((size_t)D_ * D_ * 2);
  uint16_t* wm1B = (uint16_t*)take((size_t)DHID_ * D_ * 2);
  uint16_t* wm2B = (uint16_t*)take((size_t)D_ * DHID_ * 2);
  float*    adaW = (float*)take((size_t)B_ * SIXD_ * 4);
  uint16_t* hB   = (uint16_t*)take((size_t)B_ * D_ * N_ * 2);
  uint16_t* qB   = (uint16_t*)take((size_t)B_ * D_ * N_ * 2);
  uint16_t* kvB  = (uint16_t*)take((size_t)B_ * 2 * D_ * N_ * 2);
  uint16_t* attB = (uint16_t*)take((size_t)B_ * D_ * N_ * 2);
  float*    x2   = (float*)take((size_t)B_ * D_ * N_ * 4);
  uint16_t* h2B  = (uint16_t*)take((size_t)B_ * D_ * N_ * 2);
  uint16_t* mB   = (uint16_t*)take((size_t)B_ * DHID_ * N_ * 2);

  auto cvt = [&](const float* in, uint16_t* out, int n) {
    cvt_bf16<<<(n + 255) / 256, 256, 0, stream>>>(in, out, n);
  };
  cvt(w_q,  wqB,  D_ * D_);
  cvt(w_kv, wkvB, 2 * D_ * D_);
  cvt(w_o,  woB,  D_ * D_);
  cvt(w_m1, wm1B, DHID_ * D_);
  cvt(w_m2, wm2B, D_ * DHID_);

  ada_kernel<<<dim3(SIXD_ / 256, B_), 256, 0, stream>>>(c, w_ada, b_ada, adaW);

  // h = LN(x)*(1+sc_msa)+sh_msa   (ada slices: sh=0, sc=D, g=2D, sh2=3D, sc2=4D, g2=5D)
  ln_mod<<<dim3(N_ / 32, B_), 256, 0, stream>>>(x, adaW, D_, 0, hB);

  gemm_wmma<0><<<dim3(N_ / 128, D_ / 256, B_), 256, 0, stream>>>(
      wqB, hB, b_q, qB, nullptr, nullptr, nullptr, 0, D_, D_);
  gemm_wmma<0><<<dim3(N_ / 128, (2 * D_) / 256, B_), 256, 0, stream>>>(
      wkvB, hB, b_kv, kvB, nullptr, nullptr, nullptr, 0, 2 * D_, D_);

  attn_kernel<<<dim3(N_ / 128, H_, B_), 256, 0, stream>>>(qB, kvB, attB);

  gemm_wmma<2><<<dim3(N_ / 128, D_ / 256, B_), 256, 0, stream>>>(
      woB, attB, b_o, nullptr, x2, x, adaW, 2 * D_, D_, D_);

  ln_mod<<<dim3(N_ / 32, B_), 256, 0, stream>>>(x2, adaW, 4 * D_, 3 * D_, h2B);

  gemm_wmma<1><<<dim3(N_ / 128, DHID_ / 256, B_), 256, 0, stream>>>(
      wm1B, h2B, b_m1, mB, nullptr, nullptr, nullptr, 0, DHID_, D_);

  gemm_wmma<2><<<dim3(N_ / 128, D_ / 256, B_), 256, 0, stream>>>(
      wm2B, mB, b_m2, nullptr, (float*)d_out, x2, adaW, 5 * D_, D_, DHID_);
}
